// MambaVisionMixer_27711128993876
// MI455X (gfx1250) — matrix operand order repaired and checked by reading, not compiled, not run on hardware
//
#include <hip/hip_runtime.h>
#include <math.h>

#define D_MODEL 768
#define D_INNER 1536
#define D_HALF  768
#define D_STATE 16
#define DT_RANK 48
#define DT_PAD  64          // K padded to multiple of 32 for WMMA
#define B_SZ    2
#define L_SEQ   4096
#define M_TOT   (B_SZ * L_SEQ)          // 8192
#define XD      (DT_RANK + 2 * D_STATE) // 80

typedef __attribute__((ext_vector_type(16))) __bf16 v16bf;
typedef __attribute__((ext_vector_type(8)))  __bf16 v8bf;
typedef __attribute__((ext_vector_type(8)))  float  v8f;

__device__ __forceinline__ unsigned short f2bf(float f) {
  unsigned int u = __float_as_uint(f);
  unsigned int r = u + 0x7fffu + ((u >> 16) & 1u);   // round-to-nearest-even
  return (unsigned short)(r >> 16);
}

// ---------------------------------------------------------------------------
// GEMM: C[M,N] (f32) = A[M,K] (bf16, row-major) * Bw[N,K]^T (bf16, row-major)
// One wave computes a 32x32 tile as 2x2 WMMA 16x16 tiles; operands streamed
// from global (whole working set is L2-resident on MI455X's 192MB L2).
// ---------------------------------------------------------------------------
__global__ __launch_bounds__(32)
void gemm_bf16_tn(const unsigned short* __restrict__ A,
                  const unsigned short* __restrict__ Bw,
                  float* __restrict__ C,
                  int M, int N, int K)
{
  const int lane = threadIdx.x;
  const int lrow = lane & 15;
  const int hi   = lane >> 4;              // 0: lanes 0-15, 1: lanes 16-31
  const int m0 = blockIdx.y << 5;
  const int n0 = blockIdx.x << 5;
  const bool n1ok = (n0 + 16) < N;

  const unsigned short* a0p = A  + (size_t)(m0 + lrow)      * K;
  const unsigned short* a1p = A  + (size_t)(m0 + 16 + lrow) * K;
  const unsigned short* b0p = Bw + (size_t)(n0 + lrow)      * K;
  const unsigned short* b1p = n1ok ? (Bw + (size_t)(n0 + 16 + lrow) * K) : b0p;

  const v8f zero = {0.f,0.f,0.f,0.f,0.f,0.f,0.f,0.f};
  v8f acc00 = zero, acc01 = zero, acc10 = zero, acc11 = zero;

  #pragma unroll 2
  for (int k0 = 0; k0 < K; k0 += 32) {
    // A 16x32 bf16 fragment: lanes 0-15 -> K {k0..k0+7, k0+16..k0+23},
    // lanes 16-31 -> K {k0+8..k0+15, k0+24..k0+31} (ISA 7.12.2 layout)
    const int ka = k0 + (hi << 3);
    v8bf a0lo = *(const v8bf*)(a0p + ka);
    v8bf a0hi = *(const v8bf*)(a0p + ka + 16);
    v8bf a1lo = *(const v8bf*)(a1p + ka);
    v8bf a1hi = *(const v8bf*)(a1p + ka + 16);
    v16bf af0 = __builtin_shufflevector(a0lo, a0hi, 0,1,2,3,4,5,6,7,8,9,10,11,12,13,14,15);
    v16bf af1 = __builtin_shufflevector(a1lo, a1hi, 0,1,2,3,4,5,6,7,8,9,10,11,12,13,14,15);
    // B 32x16 bf16 fragment: same K layout as A
    const int kb = k0 + (hi << 3);
    v8bf b0lo = *(const v8bf*)(b0p + kb);
    v8bf b0hi = *(const v8bf*)(b0p + kb + 16);
    v8bf b1lo = *(const v8bf*)(b1p + kb);
    v8bf b1hi = *(const v8bf*)(b1p + kb + 16);
    v16bf bf0 = __builtin_shufflevector(b0lo, b0hi, 0,1,2,3,4,5,6,7,8,9,10,11,12,13,14,15);
    v16bf bf1 = __builtin_shufflevector(b1lo, b1hi, 0,1,2,3,4,5,6,7,8,9,10,11,12,13,14,15);

    acc00 = __builtin_amdgcn_wmma_f32_16x16x32_bf16(false, af0, false, bf0, (short)0, acc00, false, false);
    acc01 = __builtin_amdgcn_wmma_f32_16x16x32_bf16(false, af0, false, bf1, (short)0, acc01, false, false);
    acc10 = __builtin_amdgcn_wmma_f32_16x16x32_bf16(false, af1, false, bf0, (short)0, acc10, false, false);
    acc11 = __builtin_amdgcn_wmma_f32_16x16x32_bf16(false, af1, false, bf1, (short)0, acc11, false, false);
  }

  // C/D layout: VGPR v holds (row m0 + v + 8*hi, col n0 + lrow)
  #pragma unroll
  for (int v = 0; v < 8; ++v) {
    const int r0 = m0 + v + (hi << 3);
    const int r1 = r0 + 16;
    const int c0 = n0 + lrow;
    C[(size_t)r0 * N + c0] = acc00[v];
    C[(size_t)r1 * N + c0] = acc10[v];
    if (n1ok) {
      C[(size_t)r0 * N + c0 + 16] = acc01[v];
      C[(size_t)r1 * N + c0 + 16] = acc11[v];
    }
  }
}

// ---------------------------------------------------------------------------
// Elementwise helpers
// ---------------------------------------------------------------------------
__global__ void f32_to_bf16_kernel(const float* __restrict__ in,
                                   unsigned short* __restrict__ out, int n) {
  int i = blockIdx.x * blockDim.x + threadIdx.x;
  if (i < n) out[i] = f2bf(in[i]);
}

// copy rows x 48 (from a row-stride `in_stride` f32 matrix) into rows x 64 bf16,
// zero-padding cols 48..63 (makes the dt GEMM K a multiple of 32)
__global__ void pad48_to_bf16_kernel(const float* __restrict__ in,
                                     unsigned short* __restrict__ out,
                                     int rows, int in_stride) {
  int i = blockIdx.x * blockDim.x + threadIdx.x;
  if (i >= rows * DT_PAD) return;
  int r = i / DT_PAD, c = i % DT_PAD;
  out[i] = (c < DT_RANK) ? f2bf(in[(size_t)r * in_stride + c]) : (unsigned short)0;
}

__global__ void softplus_bias_kernel(float* __restrict__ delta,
                                     const float* __restrict__ b_dt, int n) {
  int i = blockIdx.x * blockDim.x + threadIdx.x;
  if (i >= n) return;
  float v = delta[i] + b_dt[i % D_HALF];
  delta[i] = (v > 20.f) ? v : log1pf(__expf(v));
}

// ---------------------------------------------------------------------------
// Depthwise conv (width 4, pad (1,2)) + SiLU for both x and z halves of xz.
// Writes: x in f32 (for scan), x in bf16 (for x_dbl GEMM), and SiLU(z) in bf16
// directly into the second half of the [y,z] GEMM-A buffer.
// ---------------------------------------------------------------------------
__global__ void conv_silu_kernel(const float* __restrict__ xz,
                                 const float* __restrict__ cxw, const float* __restrict__ cxb,
                                 const float* __restrict__ czw, const float* __restrict__ czb,
                                 float* __restrict__ x_f32,
                                 unsigned short* __restrict__ x_bf16,
                                 unsigned short* __restrict__ yz_bf16)
{
  int idx = blockIdx.x * blockDim.x + threadIdx.x;
  if (idx >= M_TOT * D_HALF) return;
  const int c = idx % D_HALF;
  const int m = idx / D_HALF;       // m = b*L + t
  const int t = m % L_SEQ;

  float ax = cxb[c];
  float az = czb[c];
  #pragma unroll
  for (int j = 0; j < 4; ++j) {
    int tt = t - 1 + j;             // left pad = 1, right pad = 2
    if (tt >= 0 && tt < L_SEQ) {
      size_t src = (size_t)(m - t + tt) * D_INNER;
      ax = fmaf(cxw[c * 4 + j], xz[src + c], ax);
      az = fmaf(czw[c * 4 + j], xz[src + D_HALF + c], az);
    }
  }
  ax = ax / (1.f + __expf(-ax));    // SiLU
  az = az / (1.f + __expf(-az));
  x_f32[idx]  = ax;
  x_bf16[idx] = f2bf(ax);
  yz_bf16[(size_t)m * D_INNER + D_HALF + c] = f2bf(az);
}

// ---------------------------------------------------------------------------
// Selective scan (diagonal A). One wave = 2 channels x 16 states; lanes hold
// the state dimension, sequential over L. 16-wide shuffle reduction for y.
// Fuses the + x*D residual and writes bf16 into the [y,z] buffer first half.
// ---------------------------------------------------------------------------
__global__ __launch_bounds__(256)
void scan_kernel(const float* __restrict__ delta,
                 const float* __restrict__ x_f32,
                 const float* __restrict__ x_dbl,
                 const float* __restrict__ A_log,
                 const float* __restrict__ D_param,
                 unsigned short* __restrict__ yz_bf16)
{
  const int lane = threadIdx.x & 31;
  const int wave = threadIdx.x >> 5;
  const int s    = lane & 15;               // state index
  const int half = lane >> 4;               // which channel in the wave
  const int ch   = blockIdx.x * 16 + wave * 2 + half;   // 0..1535
  const int b    = ch / D_HALF;
  const int d    = ch % D_HALF;

  const float Aval = -__expf(A_log[d * D_STATE + s]);
  const float Dp   = D_param[d];

  const float* dptr  = delta + (size_t)b * L_SEQ * D_HALF + d;
  const float* xptr  = x_f32 + (size_t)b * L_SEQ * D_HALF + d;
  const float* bcptr = x_dbl + (size_t)b * L_SEQ * XD + DT_RANK + s;   // B at +0, C at +16
  unsigned short* yptr = yz_bf16 + (size_t)b * L_SEQ * D_INNER + d;

  float h = 0.f;
  for (int t = 0; t < L_SEQ; ++t) {
    const float dv = dptr[(size_t)t * D_HALF];
    const float xv = xptr[(size_t)t * D_HALF];
    const float Bv = bcptr[(size_t)t * XD];
    const float Cv = bcptr[(size_t)t * XD + D_STATE];
    // pull upcoming rows toward L0/L2 while the serial chain runs
    __builtin_prefetch(dptr  + (size_t)(t + 8) * D_HALF, 0, 0);
    __builtin_prefetch(bcptr + (size_t)(t + 8) * XD,     0, 0);

    const float dA = __expf(dv * Aval);
    h = fmaf(dA, h, dv * xv * Bv);

    float p = h * Cv;                       // reduce over 16 states
    p += __shfl_xor(p, 1, 16);
    p += __shfl_xor(p, 2, 16);
    p += __shfl_xor(p, 4, 16);
    p += __shfl_xor(p, 8, 16);
    if (s == 0)
      yptr[(size_t)t * D_INNER] = f2bf(p + xv * Dp);
  }
}

// ---------------------------------------------------------------------------
extern "C" void kernel_launch(void* const* d_in, const int* in_sizes, int n_in,
                              void* d_out, int out_size, void* d_ws, size_t ws_size,
                              hipStream_t stream)
{
  const float* hidden  = (const float*)d_in[0];
  const float* W_in    = (const float*)d_in[1];
  const float* cxw     = (const float*)d_in[2];
  const float* cxb     = (const float*)d_in[3];
  const float* czw     = (const float*)d_in[4];
  const float* czb     = (const float*)d_in[5];
  const float* W_xproj = (const float*)d_in[6];
  const float* W_dt    = (const float*)d_in[7];
  const float* b_dt    = (const float*)d_in[8];
  const float* A_log   = (const float*)d_in[9];
  const float* D_param = (const float*)d_in[10];
  const float* W_out   = (const float*)d_in[11];
  float* out = (float*)d_out;

  char* ws = (char*)d_ws;
  size_t off = 0;
  auto carve = [&](size_t bytes) -> void* {
    void* p = ws + off;
    off += (bytes + 255) & ~(size_t)255;
    return p;
  };
  unsigned short* hidden_bf = (unsigned short*)carve((size_t)M_TOT * D_MODEL * 2);
  unsigned short* Win_bf    = (unsigned short*)carve((size_t)D_INNER * D_MODEL * 2);
  unsigned short* Wxp_bf    = (unsigned short*)carve((size_t)XD * D_HALF * 2);
  unsigned short* Wdt_bf    = (unsigned short*)carve((size_t)D_HALF * DT_PAD * 2);
  unsigned short* Wout_bf   = (unsigned short*)carve((size_t)D_MODEL * D_INNER * 2);
  float*          xz        = (float*)        carve((size_t)M_TOT * D_INNER * 4);
  float*          x_f32     = (float*)        carve((size_t)M_TOT * D_HALF * 4);
  unsigned short* x_bf      = (unsigned short*)carve((size_t)M_TOT * D_HALF * 2);
  unsigned short* yz_bf     = (unsigned short*)carve((size_t)M_TOT * D_INNER * 2);
  float*          x_dbl     = (float*)        carve((size_t)M_TOT * XD * 4);
  unsigned short* dt_bf     = (unsigned short*)carve((size_t)M_TOT * DT_PAD * 2);
  float*          delta_f   = (float*)        carve((size_t)M_TOT * D_HALF * 4);

  const int TPB = 256;
  auto cdiv = [](int a, int b) { return (a + b - 1) / b; };

  // weight / input conversions to bf16
  f32_to_bf16_kernel<<<cdiv(M_TOT * D_MODEL, TPB), TPB, 0, stream>>>(hidden, hidden_bf, M_TOT * D_MODEL);
  f32_to_bf16_kernel<<<cdiv(D_INNER * D_MODEL, TPB), TPB, 0, stream>>>(W_in, Win_bf, D_INNER * D_MODEL);
  f32_to_bf16_kernel<<<cdiv(XD * D_HALF, TPB), TPB, 0, stream>>>(W_xproj, Wxp_bf, XD * D_HALF);
  pad48_to_bf16_kernel<<<cdiv(D_HALF * DT_PAD, TPB), TPB, 0, stream>>>(W_dt, Wdt_bf, D_HALF, DT_RANK);
  f32_to_bf16_kernel<<<cdiv(D_MODEL * D_INNER, TPB), TPB, 0, stream>>>(W_out, Wout_bf, D_MODEL * D_INNER);

  // 1) xz = hidden @ W_in^T         (8192x768 * 768x1536)
  gemm_bf16_tn<<<dim3(D_INNER / 32, M_TOT / 32), 32, 0, stream>>>(
      hidden_bf, Win_bf, xz, M_TOT, D_INNER, D_MODEL);

  // 2) depthwise conv + SiLU on x and z
  conv_silu_kernel<<<cdiv(M_TOT * D_HALF, TPB), TPB, 0, stream>>>(
      xz, cxw, cxb, czw, czb, x_f32, x_bf, yz_bf);

  // 3) x_dbl = x @ W_xproj^T        (8192x768 * 768x80)
  gemm_bf16_tn<<<dim3(cdiv(XD, 32), M_TOT / 32), 32, 0, stream>>>(
      x_bf, Wxp_bf, x_dbl, M_TOT, XD, D_HALF);

  // 4) dt (cols 0..47 of x_dbl) -> bf16 padded to K=64
  pad48_to_bf16_kernel<<<cdiv(M_TOT * DT_PAD, TPB), TPB, 0, stream>>>(
      x_dbl, dt_bf, M_TOT, XD);

  // 5) delta_raw = dt @ W_dt^T      (8192x64 * 64x768), then softplus(+b_dt)
  gemm_bf16_tn<<<dim3(D_HALF / 32, M_TOT / 32), 32, 0, stream>>>(
      dt_bf, Wdt_bf, delta_f, M_TOT, D_HALF, DT_PAD);
  softplus_bias_kernel<<<cdiv(M_TOT * D_HALF, TPB), TPB, 0, stream>>>(
      delta_f, b_dt, M_TOT * D_HALF);

  // 6) selective scan -> y (+ x*D) into first half of yz buffer
  scan_kernel<<<(B_SZ * D_HALF) / 16, 256, 0, stream>>>(
      delta_f, x_f32, x_dbl, A_log, D_param, yz_bf);

  // 7) out = [y, z] @ W_out^T       (8192x1536 * 1536x768)
  gemm_bf16_tn<<<dim3(D_MODEL / 32, M_TOT / 32), 32, 0, stream>>>(
      yz_bf, Wout_bf, out, M_TOT, D_MODEL, D_INNER);
}
